// SingleTransformLayer_91250875171388
// MI455X (gfx1250) — compile-verified, matrix operand order checked by reading, not run on hardware
//
#include <hip/hip_runtime.h>

// CDNA5 / gfx1250: wave32, WMMA 16x16x32 f16 -> f32, async global->LDS pipeline
typedef __attribute__((ext_vector_type(16))) _Float16 v16h;
typedef __attribute__((ext_vector_type(8)))  float    v8f;
typedef __attribute__((ext_vector_type(4)))  int      v4i;

#define BATCH 32768
#define DDIM  64
#define HDIM  32
#define TILE_HALFS (HDIM * DDIM)   // 2048 f16 = 4 KB per d

#if __has_builtin(__builtin_amdgcn_global_load_async_to_lds_b128)
#define HAVE_ASYNC_LDS 1
#endif

union FragB { v16h h; uint4 q[2]; };

__device__ __forceinline__ void wait_async_all() {
#if __has_builtin(__builtin_amdgcn_s_wait_asynccnt)
  __builtin_amdgcn_s_wait_asynccnt(0);
#else
  asm volatile("s_wait_asynccnt 0x0" ::: "memory");
#endif
}

#ifdef HAVE_ASYNC_LDS
__device__ __forceinline__ void async_copy_b128(const void* gsrc, void* ldst) {
  // global_load_async_to_lds_b128: 16 B per lane, memory -> LDS, ASYNCcnt-tracked
  // Builtin signature (from hipcc diagnostic): (v4i AS1*, v4i AS3*, imm offset, imm cpol)
  __builtin_amdgcn_global_load_async_to_lds_b128(
      (__attribute__((address_space(1))) v4i*)(uintptr_t)gsrc,
      (__attribute__((address_space(3))) v4i*)(unsigned int)(uintptr_t)ldst,
      0, 0);
}
#endif

// Prologue: build zero-row-inserted, transposed f16 weights WfT[d][n][k]
// (zc @ W1[d]  ==  z @ Wf[d], where Wf[d] has a zero row at k==d)
__global__ __launch_bounds__(256) void build_wft_kernel(
    const float* __restrict__ W1, _Float16* __restrict__ wft) {
  const int d = blockIdx.x;
  for (int i = threadIdx.x; i < TILE_HALFS; i += 256) {
    const int n = i >> 6;          // h index 0..31
    const int k = i & 63;          // input dim 0..63
    float v = 0.0f;
    if (k != d) {
      const int src = k - (k > d ? 1 : 0);       // undo leave-one-out
      v = W1[((size_t)d * 63 + src) * HDIM + n]; // W1[d][src][n]
    }
    wft[(size_t)d * TILE_HALFS + n * DDIM + k] = (_Float16)v;
  }
}

__global__ __launch_bounds__(256) void flow_kernel(
    const float* __restrict__ z, const _Float16* __restrict__ wft,
    const float* __restrict__ b1, const float* __restrict__ W2,
    const float* __restrict__ b2, float* __restrict__ out) {
  __shared__ _Float16 Bt[2 * TILE_HALFS];    // double-buffered 4 KB B tiles
  __shared__ float    hbuf[8 * 16 * 33];     // per-wave 16x32, stride 33 (bank pad)
  __shared__ float    xbuf[8 * 16 * 65];     // per-wave 16x64, stride 65 (bank pad)

  const int tid  = threadIdx.x;
  const int w    = tid >> 5;     // wave id 0..7
  const int lane = tid & 31;
  const int mr   = lane & 15;    // A row / C column index
  const int g    = lane >> 4;    // lane-group for WMMA K striping
  const int m0   = (blockIdx.x * 8 + w) * 16;   // first batch row of this wave

  const int hb = w * (16 * 33);
  const int xb = w * (16 * 65);

#ifdef HAVE_ASYNC_LDS
  // Kick off tile 0 before doing anything else (overlaps with A-frag build)
  async_copy_b128(((const uint4*)wft) + tid, ((uint4*)Bt) + tid);
#endif

  // ---- Preload this wave's z tile into xbuf (coalesced); stage 2 reads zd
  // from LDS and overwrites the slot with x in place. ----
  for (int idx = lane; idx < 16 * 64; idx += 32) {
    const int r = idx >> 6;
    const int c = idx & 63;
    xbuf[xb + r * 65 + c] = z[(size_t)(m0 + r) * DDIM + c];
  }

  // ---- Build A fragments once (z tile, 16 rows x K=64, f16) ----
  // 16-bit A layout: lanes 0-15 elems j<8 -> K=8g+j, j>=8 -> K=16+8g+(j-8); +32 for a1
  const float* zr = z + (size_t)(m0 + mr) * DDIM;
  float af[32];
#pragma unroll
  for (int r = 0; r < 4; ++r) {
    const float4 lo = *(const float4*)(zr + r * 16 + 8 * g);
    const float4 hi = *(const float4*)(zr + r * 16 + 8 * g + 4);
    af[r * 8 + 0] = lo.x; af[r * 8 + 1] = lo.y; af[r * 8 + 2] = lo.z; af[r * 8 + 3] = lo.w;
    af[r * 8 + 4] = hi.x; af[r * 8 + 5] = hi.y; af[r * 8 + 6] = hi.z; af[r * 8 + 7] = hi.w;
  }
  v16h a0, a1;
#pragma unroll
  for (int j = 0; j < 16; ++j) {
    a0[j] = (_Float16)af[j];
    a1[j] = (_Float16)af[16 + j];
  }

  float ldacc = 0.0f;

  for (int d = 0; d < DDIM; ++d) {
#ifdef HAVE_ASYNC_LDS
    wait_async_all();   // tile d resident in LDS
    __syncthreads();    // all waves done loading tile d / reading buf[(d+1)&1]
    if (d + 1 < DDIM)   // overlap tile d+1 load with tile d compute
      async_copy_b128(((const uint4*)(wft + (size_t)(d + 1) * TILE_HALFS)) + tid,
                      ((uint4*)(Bt + ((d + 1) & 1) * TILE_HALFS)) + tid);
#else
    __syncthreads();
    ((uint4*)(Bt + (d & 1) * TILE_HALFS))[tid] =
        ((const uint4*)(wft + (size_t)d * TILE_HALFS))[tid];
    if (d + 1 < DDIM)
      __builtin_prefetch(wft + (size_t)(d + 1) * TILE_HALFS, 0, 1);
    __syncthreads();
#endif
    const _Float16* Bcur = Bt + (d & 1) * TILE_HALFS;

    // ---- Build B fragments from LDS (ds_load_b128 x2 per operand) ----
    const int n0 = mr, n1 = 16 + mr;
    const int kA = 8 * g, kB = 16 + 8 * g, kC = 32 + 8 * g, kD = 48 + 8 * g;
    FragB b00, b01, b10, b11;
    b00.q[0] = *(const uint4*)&Bcur[n0 * DDIM + kA];
    b00.q[1] = *(const uint4*)&Bcur[n0 * DDIM + kB];
    b01.q[0] = *(const uint4*)&Bcur[n1 * DDIM + kA];
    b01.q[1] = *(const uint4*)&Bcur[n1 * DDIM + kB];
    b10.q[0] = *(const uint4*)&Bcur[n0 * DDIM + kC];
    b10.q[1] = *(const uint4*)&Bcur[n0 * DDIM + kD];
    b11.q[0] = *(const uint4*)&Bcur[n1 * DDIM + kC];
    b11.q[1] = *(const uint4*)&Bcur[n1 * DDIM + kD];

    // ---- 16x32 output tile = (16x64) @ (64x32): 4 WMMAs ----
    v8f c0 = {};
    v8f c1 = {};
    c0 = __builtin_amdgcn_wmma_f32_16x16x32_f16(false, a0, false, b00.h, (short)0, c0, false, false);
    c0 = __builtin_amdgcn_wmma_f32_16x16x32_f16(false, a1, false, b10.h, (short)0, c0, false, false);
    c1 = __builtin_amdgcn_wmma_f32_16x16x32_f16(false, a0, false, b01.h, (short)0, c1, false, false);
    c1 = __builtin_amdgcn_wmma_f32_16x16x32_f16(false, a1, false, b11.h, (short)0, c1, false, false);

    // ---- bias + relu, h -> per-wave LDS (C layout: vgpr v -> row v+8g, col mr) ----
    const float bv0 = b1[d * HDIM + mr];
    const float bv1 = b1[d * HDIM + 16 + mr];
#pragma unroll
    for (int v = 0; v < 8; ++v) {
      hbuf[hb + (v + 8 * g) * 33 + mr]      = fmaxf(c0[v] + bv0, 0.0f);
      hbuf[hb + (v + 8 * g) * 33 + 16 + mr] = fmaxf(c1[v] + bv1, 0.0f);
    }
    // No block barrier: hbuf region is wave-private and DS ops are in-order
    // within a wave (compiler inserts the s_wait_dscnt for the dependence).

    // ---- Stage 2: p = h @ W2[d] + b2[d], then sine-flow tail ----
    if (lane < 16) {
      float p0 = b2[2 * d];
      float p1 = b2[2 * d + 1];
      const float* w2d = W2 + (size_t)d * (HDIM * 2);
#pragma unroll
      for (int hh = 0; hh < HDIM; ++hh) {
        const float hv = hbuf[hb + lane * 33 + hh];
        p0 = fmaf(hv, w2d[2 * hh], p0);
        p1 = fmaf(hv, w2d[2 * hh + 1], p1);
      }
      const float zd  = xbuf[xb + lane * 65 + d];   // z preloaded in LDS
      const float e   = __expf(2.0f * p0);
      const float th  = 1.0f - 2.0f / (e + 1.0f);   // stable tanh via v_exp_f32
      const float s   = 0.95f * th;
      const float arg = zd + p1;
      const float sv  = __sinf(arg);
      const float cv  = __cosf(arg);
      xbuf[xb + lane * 65 + d] = zd + s * sv;       // overwrite z slot with x
      ldacc += __logf(fabsf(fmaf(s, cv, 1.0f)));
    }
  }

  // ---- Coalesced x write-out from padded LDS buffer (wave-private) ----
  for (int idx = lane; idx < 16 * 64; idx += 32) {
    const int r = idx >> 6;
    const int c = idx & 63;
    out[(size_t)(m0 + r) * DDIM + c] = xbuf[xb + r * 65 + c];
  }
  if (lane < 16) {
    out[(size_t)BATCH * DDIM + (m0 + lane)] = ldacc;  // log_det segment
  }
}

extern "C" void kernel_launch(void* const* d_in, const int* in_sizes, int n_in,
                              void* d_out, int out_size, void* d_ws, size_t ws_size,
                              hipStream_t stream) {
  const float* z  = (const float*)d_in[0];
  const float* W1 = (const float*)d_in[1];
  const float* b1 = (const float*)d_in[2];
  const float* W2 = (const float*)d_in[3];
  const float* b2 = (const float*)d_in[4];
  _Float16* wft = (_Float16*)d_ws;  // 64*32*64*2 = 256 KB scratch

  build_wft_kernel<<<DDIM, 256, 0, stream>>>(W1, wft);
  flow_kernel<<<BATCH / 128, 256, 0, stream>>>(z, wft, b1, W2, b2, (float*)d_out);
}
